// SelfAttnV3_63118839382026
// MI455X (gfx1250) — matrix operand order repaired and checked by reading, not compiled, not run on hardware
//
#include <hip/hip_runtime.h>
#include <hip/hip_bf16.h>
#include <math.h>

#define BB 4
#define SS 2048
#define DD 1024
#define TD 3072
#define LDB_STRIDE 72   // 64 + 8 pad: de-correlates LDS banks (36-bank stride)

typedef __attribute__((ext_vector_type(16))) __bf16 v16bf;
typedef __attribute__((ext_vector_type(8)))  __bf16 bfx8;
typedef __attribute__((ext_vector_type(4)))  __bf16 bfx4;
typedef __attribute__((ext_vector_type(8)))  float  v8f;
typedef int sa_v4i __attribute__((vector_size(16)));

// ---- CDNA5 async global->LDS path --------------------------------------
#if __has_builtin(__builtin_amdgcn_global_load_async_to_lds_b128)
#define SA_ASYNC 1
#else
#define SA_ASYNC 0
#endif

#if __has_builtin(__builtin_amdgcn_s_wait_asynccnt)
#define SA_WAITA(n) __builtin_amdgcn_s_wait_asynccnt(n)
#elif SA_ASYNC
#define SA_WAITA(n) asm volatile("s_wait_asynccnt %0" ::"i"(n) : "memory")
#else
#define SA_WAITA(n) ((void)0)
#endif

static __device__ __forceinline__ void async_copy16(const __bf16* g,
                                                    __bf16* l) {
#if SA_ASYNC
  __builtin_amdgcn_global_load_async_to_lds_b128(
      (__attribute__((address_space(1))) sa_v4i*)g,
      (__attribute__((address_space(3))) sa_v4i*)l, 0, 0);
#else
  *(int4*)l = *(const int4*)g;   // synchronous fallback (still via LDS)
#endif
}

// ---- WMMA helpers (CDNA5 16x16x32 bf16, fp32 accum) --------------------
static __device__ __forceinline__ v8f wmma_bf16(v16bf a, v16bf b, v8f c) {
  return __builtin_amdgcn_wmma_f32_16x16x32_bf16(false, a, false, b,
                                                 (short)0, c, false, false);
}

// A-fragment: lane half h=0 holds K={kk..kk+15},
//             h=1 holds K={kk+16..kk+31}  (two b128 loads)
static __device__ __forceinline__ v16bf ldA(const __bf16* __restrict__ row,
                                            int kk, int h) {
  bfx8 lo = *(const bfx8*)(row + kk + 16 * h);
  bfx8 hi = *(const bfx8*)(row + kk + 16 * h + 8);
  return __builtin_shufflevector(lo, hi, 0, 1, 2, 3, 4, 5, 6, 7,
                                 8, 9, 10, 11, 12, 13, 14, 15);
}

// B-fragment from a contiguous 32B run, but only 16B alignment guaranteed
static __device__ __forceinline__ v16bf ldB16(const __bf16* p) {
  bfx8 lo = *(const bfx8*)p;
  bfx8 hi = *(const bfx8*)(p + 8);
  return __builtin_shufflevector(lo, hi, 0, 1, 2, 3, 4, 5, 6, 7,
                                 8, 9, 10, 11, 12, 13, 14, 15);
}

// Stage a 64-row x 64-col bf16 tile (rows of gbase) into LDS, 512B/wave/instr
static __device__ __forceinline__ void stage_tile(
    const __bf16* __restrict__ gbase, size_t grs, int n0, int kc,
    __bf16* lbuf, int tid) {
#pragma unroll
  for (int r = 0; r < 4; ++r) {
    int flat = r * 128 + tid;
    int row = flat >> 3, seg = flat & 7;
    async_copy16(gbase + (size_t)(n0 + row) * grs + kc + seg * 8,
                 lbuf + row * LDB_STRIDE + seg * 8);
  }
}

// ---- Kernel 0a: x fp32 -> bf16 -----------------------------------------
__global__ __launch_bounds__(256) void sa_cvt_x(const float* __restrict__ x,
                                                __bf16* __restrict__ xb,
                                                int n) {
  int i = (blockIdx.x * 256 + threadIdx.x) * 4;
  if (i + 3 < n) {
    float4 v = *(const float4*)(x + i);
    bfx4 o;
    o[0] = (__bf16)v.x; o[1] = (__bf16)v.y;
    o[2] = (__bf16)v.z; o[3] = (__bf16)v.w;
    *(bfx4*)(xb + i) = o;
  }
}

// ---- Kernel 0b: W [D,3D] fp32 -> W^T [3D,D] bf16 (LDS tile transpose) ---
__global__ __launch_bounds__(256) void sa_tr_w(const float* __restrict__ w,
                                               __bf16* __restrict__ wt) {
  __shared__ __bf16 tile[32][33];
  int j0 = blockIdx.x * 32;
  int i0 = blockIdx.y * 32;
  int tj = threadIdx.x & 31, tg = threadIdx.x >> 5;
#pragma unroll
  for (int rr = 0; rr < 4; ++rr) {
    int ti = tg + rr * 8;
    tile[ti][tj] = (__bf16)w[(size_t)(i0 + ti) * TD + j0 + tj];
  }
  __syncthreads();
  int ti = threadIdx.x & 31, sg = threadIdx.x >> 5;
#pragma unroll
  for (int rr = 0; rr < 4; ++rr) {
    int tj2 = sg + rr * 8;
    wt[(size_t)(j0 + tj2) * DD + i0 + ti] = tile[ti][tj2];
  }
}

// ---- Kernel 1: QKV = X * W ---------------------------------------------
// 4 waves/block; block = 128x64 output tile, wave = 32x64 strip (8 accs).
// B tile (W^T rows) double-buffered into LDS with async copies.
__global__ __launch_bounds__(128) void sa_qkv_gemm(
    const __bf16* __restrict__ xb,   // [B*S, D]
    const __bf16* __restrict__ wt,   // [3D, D]  (= W^T)
    __bf16* __restrict__ Q,          // [B*S, D]
    __bf16* __restrict__ K,          // [B*S, D]
    __bf16* __restrict__ Vt) {       // [B, D, S]
  __shared__ __bf16 bt[2][64 * LDB_STRIDE];
  int tid = threadIdx.x, wave = tid >> 5, lane = tid & 31;
  int l = lane & 15, h = lane >> 4;
  int n0 = blockIdx.x * 64;
  int m0 = blockIdx.y * 128 + wave * 32;
  v8f acc[2][4] = {};
  const __bf16* arow0 = xb + (size_t)(m0 + l) * DD;
  const __bf16* arow1 = xb + (size_t)(m0 + 16 + l) * DD;

  const int NC = DD / 64;  // 16 K-chunks
  stage_tile(wt, DD, n0, 0, bt[0], tid);
  for (int c = 0; c < NC; ++c) {
    int buf = c & 1;
    if (c + 1 < NC) {
      stage_tile(wt, DD, n0, (c + 1) * 64, bt[buf ^ 1], tid);
      SA_WAITA(4);   // previous chunk's 4 async ops done; next 4 in flight
    } else {
      SA_WAITA(0);
    }
    __syncthreads();
    int kc = c * 64;
#pragma unroll
    for (int kk2 = 0; kk2 < 64; kk2 += 32) {
      v16bf a0 = ldA(arow0, kc + kk2, h);
      v16bf a1 = ldA(arow1, kc + kk2, h);
#pragma unroll
      for (int t = 0; t < 4; ++t) {
        v16bf bf = ldB16(&bt[buf][(t * 16 + l) * LDB_STRIDE + kk2 + 16 * h]);
        acc[0][t] = wmma_bf16(a0, bf, acc[0][t]);
        acc[1][t] = wmma_bf16(a1, bf, acc[1][t]);
      }
    }
    __syncthreads();  // protect buf before it is restaged two chunks later
  }

#pragma unroll
  for (int g = 0; g < 2; ++g) {
    int mg = m0 + g * 16;
    int bidx = mg >> 11;
    int s0 = (mg & (SS - 1));
#pragma unroll
    for (int t = 0; t < 4; ++t) {
      int gn = n0 + t * 16;
      int third = gn >> 10;  // 0=Q, 1=K, 2=V
      int lc = (gn & (DD - 1)) + l;
      if (third == 2) {
        bfx8 vv;
#pragma unroll
        for (int r = 0; r < 8; ++r) vv[r] = (__bf16)acc[g][t][r];
        *(bfx8*)(Vt + ((size_t)bidx * DD + lc) * SS + s0 + 8 * h) = vv;
      } else {
        __bf16* Ob = (third == 0) ? Q : K;
#pragma unroll
        for (int r = 0; r < 8; ++r)
          Ob[(size_t)(mg + r + 8 * h) * DD + lc] = (__bf16)acc[g][t][r];
      }
    }
  }
}

// ---- Kernel 2: scores (QK^T, causal) + softmax -> P bf16 ---------------
// 32 query rows per block; 32x2048 fp32 scores live in 256 KB of CDNA5 LDS.
// Each K-tile B-fragment is reused for two query row-tiles.
__global__ __launch_bounds__(256) void sa_attn_softmax(
    const __bf16* __restrict__ Q,    // [B, S, D]
    const __bf16* __restrict__ Km,   // [B, S, D]
    __bf16* __restrict__ P) {        // [B, S, S]
  extern __shared__ __align__(16) char smem_raw[];
  float* sc = (float*)smem_raw;      // [32][2048] = 256 KB
  float* red = sc + 32 * SS;         // [32][8]
  float* red2 = red + 256;           // [32][8]

  int qb = blockIdx.x, b = blockIdx.y;   // 32-row query block
  int q0 = qb * 32;
  int qtop = qb * 2 + 1;                 // highest 16-row query tile index
  int tid = threadIdx.x, wave = tid >> 5, lane = tid & 31;
  int l = lane & 15, h = lane >> 4;
  const __bf16* Qb = Q + (size_t)b * SS * DD;
  const __bf16* Kb = Km + (size_t)b * SS * DD;
  const __bf16* arow0 = Qb + (size_t)(q0 + l) * DD;
  const __bf16* arow1 = Qb + (size_t)(q0 + 16 + l) * DD;

  for (int kt = wave; kt <= qtop; kt += 8) {
    v8f acc0 = {}, acc1 = {};
    const __bf16* brow = Kb + (size_t)(kt * 16 + l) * DD;
#pragma unroll 4
    for (int kk = 0; kk < DD; kk += 32) {
      v16bf bf = ldB16(brow + kk + 16 * h);
      acc0 = wmma_bf16(ldA(arow0, kk, h), bf, acc0);
      acc1 = wmma_bf16(ldA(arow1, kk, h), bf, acc1);
    }
#pragma unroll
    for (int r = 0; r < 8; ++r) {
      sc[(r + 8 * h) * SS + kt * 16 + l] = acc0[r] * 0.03125f;       // 1/32
      sc[(16 + r + 8 * h) * SS + kt * 16 + l] = acc1[r] * 0.03125f;
    }
  }
  __syncthreads();

  // thread (r, seg): row r (0..31), cols k = seg + 8*i  (bank-spread)
  int r = tid >> 3, seg = tid & 7;
  int q = q0 + r;
  float mx = -1e30f;
  for (int i = 0; i < 256; ++i) {
    int k = seg + i * 8;
    if (k <= q) mx = fmaxf(mx, sc[r * SS + k]);
  }
  red[r * 8 + seg] = mx;
  __syncthreads();
  float m = -1e30f;
#pragma unroll
  for (int j = 0; j < 8; ++j) m = fmaxf(m, red[r * 8 + j]);
  float ssum = 0.f;
  for (int i = 0; i < 256; ++i) {
    int k = seg + i * 8;
    if (k <= q) {
      float e = __expf(sc[r * SS + k] - m);
      sc[r * SS + k] = e;
      ssum += e;
    }
  }
  red2[r * 8 + seg] = ssum;
  __syncthreads();
  float tot = 0.f;
#pragma unroll
  for (int j = 0; j < 8; ++j) tot += red2[r * 8 + j];
  float inv = 1.f / tot;
  __bf16* Pr = P + ((size_t)b * SS + q) * SS;
  for (int i = 0; i < 256; ++i) {
    int k = seg + i * 8;
    float v = (k <= q) ? sc[r * SS + k] * inv : 0.f;
    Pr[k] = (__bf16)v;
  }
}

// ---- Kernel 3: out = P * V (causal-truncated) --------------------------
// 4 waves/block; block = 128x64 output tile, wave = 32x64 strip.
__global__ __launch_bounds__(128) void sa_pv_gemm(
    const __bf16* __restrict__ P,    // [B, S, S]
    const __bf16* __restrict__ Vt,   // [B, D, S]
    float* __restrict__ out) {       // [B, S, D]
  __shared__ __bf16 bt[2][64 * LDB_STRIDE];
  int tid = threadIdx.x, wave = tid >> 5, lane = tid & 31;
  int l = lane & 15, h = lane >> 4;
  int b = blockIdx.z, n0 = blockIdx.x * 64;
  int m0 = blockIdx.y * 128 + wave * 32;
  const __bf16* Pb = P + (size_t)b * SS * SS;
  const __bf16* Vb = Vt + (size_t)b * DD * SS;
  const __bf16* arow0 = Pb + (size_t)(m0 + l) * SS;
  const __bf16* arow1 = Pb + (size_t)(m0 + 16 + l) * SS;
  v8f acc[2][4] = {};

  const int NC = (blockIdx.y + 1) * 2;  // causal: K up to row block's edge
  stage_tile(Vb, SS, n0, 0, bt[0], tid);
  for (int c = 0; c < NC; ++c) {
    int buf = c & 1;
    if (c + 1 < NC) {
      stage_tile(Vb, SS, n0, (c + 1) * 64, bt[buf ^ 1], tid);
      SA_WAITA(4);
    } else {
      SA_WAITA(0);
    }
    __syncthreads();
    int kc = c * 64;
#pragma unroll
    for (int kk2 = 0; kk2 < 64; kk2 += 32) {
      v16bf a0 = ldA(arow0, kc + kk2, h);
      v16bf a1 = ldA(arow1, kc + kk2, h);
#pragma unroll
      for (int t = 0; t < 4; ++t) {
        v16bf bf = ldB16(&bt[buf][(t * 16 + l) * LDB_STRIDE + kk2 + 16 * h]);
        acc[0][t] = wmma_bf16(a0, bf, acc[0][t]);
        acc[1][t] = wmma_bf16(a1, bf, acc[1][t]);
      }
    }
    __syncthreads();
  }

  float* Ob = out + (size_t)b * SS * DD;
#pragma unroll
  for (int g = 0; g < 2; ++g)
#pragma unroll
    for (int t = 0; t < 4; ++t)
#pragma unroll
      for (int r = 0; r < 8; ++r)
        Ob[(size_t)(m0 + g * 16 + r + 8 * h) * DD + n0 + t * 16 + l] =
            acc[g][t][r];
}

// ---- Host launch --------------------------------------------------------
extern "C" void kernel_launch(void* const* d_in, const int* in_sizes, int n_in,
                              void* d_out, int out_size, void* d_ws,
                              size_t ws_size, hipStream_t stream) {
  const float* x = (const float*)d_in[0];   // [B,S,D]
  const float* W = (const float*)d_in[1];   // [D,3D]
  float* out = (float*)d_out;
  char* ws = (char*)d_ws;

  const size_t xbB = (size_t)BB * SS * DD * 2;       // 16 MB
  const size_t wtB = (size_t)TD * DD * 2;            //  6 MB
  const size_t qB  = (size_t)BB * SS * DD * 2;       // 16 MB
  __bf16* Xb = (__bf16*)(ws);
  __bf16* Wt = (__bf16*)(ws + xbB);
  __bf16* Q  = (__bf16*)(ws + xbB + wtB);
  __bf16* K  = (__bf16*)(ws + xbB + wtB + qB);
  __bf16* Vt = (__bf16*)(ws + xbB + wtB + 2 * qB);
  __bf16* P  = (__bf16*)(ws + xbB + wtB + 3 * qB);   // 32 MB

  int nX = BB * SS * DD;
  sa_cvt_x<<<nX / (256 * 4), 256, 0, stream>>>(x, Xb, nX);
  sa_tr_w<<<dim3(TD / 32, DD / 32), 256, 0, stream>>>(W, Wt);
  sa_qkv_gemm<<<dim3(TD / 64, BB * SS / 128), 128, 0, stream>>>(Xb, Wt, Q, K,
                                                                Vt);
  size_t smem = (size_t)(32 * SS + 512) * sizeof(float);  // 258 KB < 320 KB
  sa_attn_softmax<<<dim3(SS / 32, BB), 256, smem, stream>>>(Q, K, P);
  sa_pv_gemm<<<dim3(DD / 64, SS / 128, BB), 128, 0, stream>>>(P, Vt, out);
}
